// HippocampalFastWeight_73126113182075
// MI455X (gfx1250) — matrix operand order repaired and checked by reading, not compiled, not run on hardware
//
#include <hip/hip_runtime.h>
#include <hip/hip_bf16.h>

// ---------------------------------------------------------------------------
// HippocampalFastWeight for MI455X (gfx1250, wave32, WMMA).
//
// All GEMM-shaped einsums are lowered to v_wmma_f32_16x16x32_bf16 with
// double-buffered async Global->LDS staging (GLOBAL_LOAD_ASYNC_TO_LDS_B128,
// ASYNCcnt / s_wait_asynccnt) so tile movement overlaps WMMA issue and
// operands are shared across the 8 waves of a block.
// Problem dims: B=8, T=2048, D=512, N=2048, K_CA3=64.
// Workspace need: ~625 MB (carved linearly from d_ws).
// ---------------------------------------------------------------------------

typedef __bf16        v16bf __attribute__((ext_vector_type(16)));
typedef float         v8f   __attribute__((ext_vector_type(8)));
typedef unsigned int  u32x4 __attribute__((ext_vector_type(4)));

__device__ __forceinline__ unsigned short f32_to_bf16u(float f) {
    unsigned int u = __float_as_uint(f);
    u += 0x7fffu + ((u >> 16) & 1u);   // round-to-nearest-even
    return (unsigned short)(u >> 16);
}

union FragAB {
    v16bf v;
    u32x4 q[2];
};

// one 16-byte per-lane async copy: global -> LDS, tracked by ASYNCcnt
__device__ __forceinline__ void async_copy_b128(unsigned int lds_off,
                                                const void* gptr) {
    unsigned long long ga = (unsigned long long)(size_t)gptr;
    asm volatile("global_load_async_to_lds_b128 %0, %1, off"
                 :: "v"(lds_off), "v"(ga) : "memory");
}

__device__ __forceinline__ void wait_async0() {
#if defined(__has_builtin)
#if __has_builtin(__builtin_amdgcn_s_wait_asynccnt)
    __builtin_amdgcn_s_wait_asynccnt(0);
#else
    asm volatile("s_wait_asynccnt 0x0" ::: "memory");
#endif
#else
    asm volatile("s_wait_asynccnt 0x0" ::: "memory");
#endif
}

__device__ __forceinline__ unsigned int lds_offset_of(const void* p) {
    // generic pointer to LDS keeps the byte offset in the low 32 bits
    return (unsigned int)(size_t)p;
}

// ---------------------------------------------------------------------------
// Generic batched GEMM:  C[M x Ncol] = A[M x K] * B[Ncol x K]^T   (TN form)
// A, B are bf16 (as ushort), K-major (lda/ldb = K stride in elements).
// C is f32 row-major. Block tile 64x128, 8 waves, wave tile 32x32
// (2x2 WMMA 16x16x32 bf16 accumulators). M%64==0, Ncol%128==0, K%32==0.
//
// Pipeline per 32-K step:
//   s_wait_asynccnt 0 ; s_barrier      (copies from previous step done)
//   issue 768 async b128 copies (A 4KB + B 8KB) into buf^1
//   8 waves read WMMA fragments from LDS (ds_load_b128) per ISA 7.12.2:
//     A 16x32: lanes 0-15 K {0..7,16..23}, lanes 16-31 K {8..15,24..31}
//     B 32x16: lanes 0-15 K {0..7,16..23}, lanes 16-31 K {8..15,24..31}
// ---------------------------------------------------------------------------
__global__ __launch_bounds__(256) void gemm_bf16_wmma(
    const unsigned short* __restrict__ A,
    const unsigned short* __restrict__ Bm,
    float* __restrict__ C,
    int M, int Ncol, int K, int lda, int ldb, int ldc,
    long long strideA, long long strideB, long long strideC)
{
    __shared__ unsigned short sA[2][64 * 32];    // 2 x 4 KB
    __shared__ unsigned short sB[2][128 * 32];   // 2 x 8 KB

    const long long bz = blockIdx.z;
    A  += bz * strideA;
    Bm += bz * strideB;
    C  += bz * strideC;

    const int tid  = threadIdx.x;
    const int lane = tid & 31;
    const int wave = tid >> 5;
    const int wm = wave & 1;        // 2 waves along M  -> 64 rows
    const int wn = wave >> 1;       // 4 waves along N  -> 128 cols

    const int rowBlk = blockIdx.x * 64;
    const int colBlk = blockIdx.y * 128;

    // cooperative copy slots: one A chunk + two B chunks per thread (16B each)
    const int cpRow = tid >> 2;           // 0..63
    const int cpSeg = (tid & 3) * 8;      // elements within the 32-K strip
    const unsigned short* gA  = A  + (long long)(rowBlk + cpRow) * lda + cpSeg;
    const unsigned short* gB0 = Bm + (long long)(colBlk + cpRow) * ldb + cpSeg;
    const unsigned short* gB1 = Bm + (long long)(colBlk + 64 + cpRow) * ldb + cpSeg;

    const unsigned int ldsA  = lds_offset_of(&sA[0][cpRow * 32 + cpSeg]);
    const unsigned int ldsB0 = lds_offset_of(&sB[0][cpRow * 32 + cpSeg]);
    const unsigned int ldsB1 = lds_offset_of(&sB[0][(64 + cpRow) * 32 + cpSeg]);
    const unsigned int bufStrideA = 64 * 32 * 2;    // bytes between buffers
    const unsigned int bufStrideB = 128 * 32 * 2;

    // per-wave fragment coordinates inside the LDS tiles
    const int aRowL = wm * 32 + (lane & 15);
    const int aOff  = (lane >> 4) ? 8 : 0;
    const int bColL = wn * 32 + (lane & 15);
    const int bOff  = (lane >> 4) ? 8 : 0;

    v8f acc00 = {}, acc01 = {}, acc10 = {}, acc11 = {};

    // prologue: stage K tile 0 into buffer 0
    async_copy_b128(ldsA, gA);
    async_copy_b128(ldsB0, gB0);
    async_copy_b128(ldsB1, gB1);

    int cur = 0;
    for (int kk = 0; kk < K; kk += 32) {
        wait_async0();          // this wave's staged copies have landed in LDS
        __syncthreads();        // -> whole block's copies have landed

        if (kk + 32 < K) {      // stage next K tile into the other buffer
            const int nb = cur ^ 1;
            async_copy_b128(ldsA + nb * bufStrideA, gA + kk + 32);
            async_copy_b128(ldsB0 + nb * bufStrideB, gB0 + kk + 32);
            async_copy_b128(ldsB1 + nb * bufStrideB, gB1 + kk + 32);
            // warm L2 for the K+2 tile (global_prefetch_b8)
            __builtin_prefetch(gA + kk + 64, 0, 2);
            __builtin_prefetch(gB0 + kk + 64, 0, 2);
            __builtin_prefetch(gB1 + kk + 64, 0, 2);
        }

        const unsigned short* bA = sA[cur];
        const unsigned short* bB = sB[cur];
        FragAB a0, a1, b0, b1;
        a0.q[0] = *(const u32x4*)&bA[aRowL * 32 + aOff];
        a0.q[1] = *(const u32x4*)&bA[aRowL * 32 + aOff + 16];
        a1.q[0] = *(const u32x4*)&bA[(aRowL + 16) * 32 + aOff];
        a1.q[1] = *(const u32x4*)&bA[(aRowL + 16) * 32 + aOff + 16];
        b0.q[0] = *(const u32x4*)&bB[bColL * 32 + bOff];
        b0.q[1] = *(const u32x4*)&bB[bColL * 32 + bOff + 16];
        b1.q[0] = *(const u32x4*)&bB[(bColL + 16) * 32 + bOff];
        b1.q[1] = *(const u32x4*)&bB[(bColL + 16) * 32 + bOff + 16];

        acc00 = __builtin_amdgcn_wmma_f32_16x16x32_bf16(
            false, a0.v, false, b0.v, (short)0, acc00, false, false);
        acc01 = __builtin_amdgcn_wmma_f32_16x16x32_bf16(
            false, a0.v, false, b1.v, (short)0, acc01, false, false);
        acc10 = __builtin_amdgcn_wmma_f32_16x16x32_bf16(
            false, a1.v, false, b0.v, (short)0, acc10, false, false);
        acc11 = __builtin_amdgcn_wmma_f32_16x16x32_bf16(
            false, a1.v, false, b1.v, (short)0, acc11, false, false);

        __syncthreads();        // all waves done reading buf `cur`
        cur ^= 1;
    }

    // C/D layout: VGPR r, lanes 0-15 -> row r, lanes 16-31 -> row r+8
    const int cCol = colBlk + wn * 32 + (lane & 15);
    const int cRow = rowBlk + wm * 32 + ((lane >> 4) ? 8 : 0);
#pragma unroll
    for (int r = 0; r < 8; ++r) {
        C[(long long)(cRow + r) * ldc + cCol]           = acc00[r];
        C[(long long)(cRow + r) * ldc + cCol + 16]      = acc01[r];
        C[(long long)(cRow + r + 16) * ldc + cCol]      = acc10[r];
        C[(long long)(cRow + r + 16) * ldc + cCol + 16] = acc11[r];
    }
}

// ---------------------------------------------------------------------------
// relu -> kWTA(top-64) -> L2 normalize, one 256-thread block per row of N=2048.
// Emits dense bf16 row (zeros outside the top-64 support).
// ---------------------------------------------------------------------------
#define SP_N 2048
#define SP_K 64

__global__ __launch_bounds__(256) void sparsify_kernel(
    const float* __restrict__ In, unsigned short* __restrict__ OutBf)
{
    __shared__ float vals[SP_N];
    __shared__ float red[256];
    __shared__ int   redi[256];
    __shared__ float invn;

    const long long row = blockIdx.x;
    const int tid = threadIdx.x;
    const float* in = In + row * SP_N;

    for (int i = tid; i < SP_N; i += 256) {
        float v = in[i];
        vals[i] = v > 0.f ? v : 0.f;               // relu
    }
    __syncthreads();

    // 64 sequential argmax extractions; selected entry v is marked as (-v-2),
    // which is < any remaining relu'd value and recoverable.
    for (int it = 0; it < SP_K; ++it) {
        float m = -1e30f; int mi = 0;
        for (int i = tid; i < SP_N; i += 256) {
            float v = vals[i];
            if (v > m) { m = v; mi = i; }
        }
        red[tid] = m; redi[tid] = mi;
        __syncthreads();
        for (int s = 128; s > 0; s >>= 1) {
            if (tid < s && red[tid + s] > red[tid]) {
                red[tid] = red[tid + s]; redi[tid] = redi[tid + s];
            }
            __syncthreads();
        }
        if (tid == 0) {
            int sel = redi[0];
            vals[sel] = -vals[sel] - 2.0f;
        }
        __syncthreads();
    }

    // L2 norm over the selected 64 entries
    float s = 0.f;
    for (int i = tid; i < SP_N; i += 256) {
        float v = vals[i];
        if (v < -0.5f) { float o = -v - 2.0f; s += o * o; }
    }
    red[tid] = s;
    __syncthreads();
    for (int st = 128; st > 0; st >>= 1) {
        if (tid < st) red[tid] += red[tid + st];
        __syncthreads();
    }
    if (tid == 0) invn = 1.0f / fmaxf(sqrtf(red[0]), 1e-10f);
    __syncthreads();

    const float inv = invn;
    unsigned short* out = OutBf + row * SP_N;
    for (int i = tid; i < SP_N; i += 256) {
        float v = vals[i];
        float o = (v < -0.5f) ? (-v - 2.0f) * inv : 0.f;
        out[i] = f32_to_bf16u(o);
    }
}

// ---------------------------------------------------------------------------
// sd (B,T,N) -> prevT[b][n][t] = sd[b][t][n] ; nextT[b][n][t] = sd[b][t+1][n]
// (zero-padded at t = T-1). Keeping two buffers keeps the Hebbian GEMM's
// K pointers 16-byte aligned (a +1-element shift would break b128 loads).
// ---------------------------------------------------------------------------
__global__ void transpose_shift_kernel(
    const unsigned short* __restrict__ sd,
    unsigned short* __restrict__ prevT,
    unsigned short* __restrict__ nextT,
    int T, int N)
{
    const long long tot = (long long)T * N;
    const long long idx = blockIdx.x * 256LL + threadIdx.x;
    const long long b = blockIdx.y;
    if (idx >= tot) return;
    const int t = (int)(idx / N);
    const int n = (int)(idx % N);
    const unsigned short v = sd[b * tot + idx];
    prevT[b * tot + (long long)n * T + t] = v;
    if (t > 0)     nextT[b * tot + (long long)n * T + (t - 1)] = v;
    if (t == T - 1) nextT[b * tot + (long long)n * T + (T - 1)] = 0;
}

// W (B,N,N) f32 -> bf16 with zeroed diagonal
__global__ void mask_convert_w_kernel(
    const float* __restrict__ W, unsigned short* __restrict__ Wb, int N)
{
    const long long tot = (long long)N * N;
    const long long idx = blockIdx.x * 256LL + threadIdx.x;
    const long long b = blockIdx.y;
    if (idx >= tot) return;
    const int n = (int)(idx / N);
    const int m = (int)(idx % N);
    float v = (n == m) ? 0.f : W[b * tot + idx];
    Wb[b * tot + idx] = f32_to_bf16u(v);
}

__global__ void convert_f32_bf16_kernel(
    const float* __restrict__ in, unsigned short* __restrict__ out, long long n)
{
    const long long i = blockIdx.x * 256LL + threadIdx.x;
    if (i < n) out[i] = f32_to_bf16u(in[i]);
}

// out = x + sigmoid(gate_logit + gate_b) * pred
__global__ void final_gate_kernel(
    const float* __restrict__ x, const float* __restrict__ gl,
    const float* __restrict__ pred, const float* __restrict__ gate_b,
    float* __restrict__ out, long long total, int D)
{
    const long long i = blockIdx.x * 256LL + threadIdx.x;
    if (i >= total) return;
    const int d = (int)(i % D);
    const float g = 1.f / (1.f + __expf(-(gl[i] + gate_b[d])));
    out[i] = x[i] + g * pred[i];
}

// ---------------------------------------------------------------------------
extern "C" void kernel_launch(void* const* d_in, const int* in_sizes, int n_in,
                              void* d_out, int out_size, void* d_ws, size_t ws_size,
                              hipStream_t stream)
{
    (void)in_sizes; (void)n_in; (void)out_size; (void)ws_size;

    const float* x      = (const float*)d_in[0];  // (8,2048,512)
    const float* down_W = (const float*)d_in[1];  // (2048,512)
    const float* up_W   = (const float*)d_in[2];  // (512,2048)
    const float* gate_W = (const float*)d_in[3];  // (512,512)
    const float* gate_b = (const float*)d_in[4];  // (512,)
    float* out = (float*)d_out;

    const int Bb = 8, T = 2048, D = 512, N = 2048;
    const long long M = (long long)Bb * T;          // 16384 rows
    const long long TN = (long long)T * N;          // per-batch sd size
    const long long NN = (long long)N * N;          // per-batch W size

    // ---- carve workspace (256B aligned chunks) ----
    char* p = (char*)d_ws;
    auto carve = [&](size_t bytes) -> char* {
        char* r = p;
        p += (bytes + 255) & ~(size_t)255;
        return r;
    };
    unsigned short* xb  = (unsigned short*)carve(M * D * 2);        // x bf16
    unsigned short* dwb = (unsigned short*)carve((size_t)N * D * 2);// down_W bf16
    unsigned short* uwb = (unsigned short*)carve((size_t)D * N * 2);// up_W bf16
    unsigned short* gwb = (unsigned short*)carve((size_t)D * D * 2);// gate_W bf16
    float*          h   = (float*)carve(M * N * 4);                 // h / succ (reused)
    unsigned short* sd  = (unsigned short*)carve(M * N * 2);        // sparse_all bf16
    unsigned short* sdP = (unsigned short*)carve(Bb * TN * 2);      // prevT / succ_dense (reused)
    unsigned short* sdX = (unsigned short*)carve(Bb * TN * 2);      // nextT
    float*          W   = (float*)carve(Bb * NN * 4);               // Hebbian W f32
    unsigned short* Wb  = (unsigned short*)carve(Bb * NN * 2);      // W bf16 (diag masked)
    float*          pr  = (float*)carve(M * D * 4);                 // prediction
    float*          gl  = (float*)carve(M * D * 4);                 // gate logits

    // ---- 1) convert inputs to bf16 ----
    {
        long long n;
        n = M * D;
        convert_f32_bf16_kernel<<<dim3((unsigned)((n + 255) / 256)), 256, 0, stream>>>(x, xb, n);
        n = (long long)N * D;
        convert_f32_bf16_kernel<<<dim3((unsigned)((n + 255) / 256)), 256, 0, stream>>>(down_W, dwb, n);
        n = (long long)D * N;
        convert_f32_bf16_kernel<<<dim3((unsigned)((n + 255) / 256)), 256, 0, stream>>>(up_W, uwb, n);
        n = (long long)D * D;
        convert_f32_bf16_kernel<<<dim3((unsigned)((n + 255) / 256)), 256, 0, stream>>>(gate_W, gwb, n);
    }

    // ---- 2) h = x . down_W^T   (16384 x 2048, K=512) ----
    gemm_bf16_wmma<<<dim3(M / 64, N / 128, 1), 256, 0, stream>>>(
        xb, dwb, h, (int)M, N, D, D, D, N, 0, 0, 0);

    // ---- 3) gate logits = x . gate_W^T  (16384 x 512, K=512) ----
    gemm_bf16_wmma<<<dim3(M / 64, D / 128, 1), 256, 0, stream>>>(
        xb, gwb, gl, (int)M, D, D, D, D, D, 0, 0, 0);

    // ---- 4) sparse_all = l2norm(kwta(relu(h))) ----
    sparsify_kernel<<<dim3((unsigned)M), 256, 0, stream>>>(h, sd);

    // ---- 5) transposed / shifted copies for the Hebbian GEMM ----
    transpose_shift_kernel<<<dim3((unsigned)((TN + 255) / 256), Bb), 256, 0, stream>>>(
        sd, sdP, sdX, T, N);

    // ---- 6) W[b] = nextT . prevT^T  (2048 x 2048, K=T) batched over 8 ----
    gemm_bf16_wmma<<<dim3(N / 64, N / 128, Bb), 256, 0, stream>>>(
        sdX, sdP, W, N, N, T, T, T, N, TN, TN, NN);

    // ---- 7) zero diagonal + convert to bf16 ----
    mask_convert_w_kernel<<<dim3((unsigned)((NN + 255) / 256), Bb), 256, 0, stream>>>(
        W, Wb, N);

    // ---- 8) succ = sd . W^T per batch (T x N, K=N), reuse h buffer ----
    gemm_bf16_wmma<<<dim3(T / 64, N / 128, Bb), 256, 0, stream>>>(
        sd, Wb, h, T, N, N, N, N, N, TN, NN, TN);

    // ---- 9) sparsify succ -> dense bf16 (reuse prevT buffer) ----
    sparsify_kernel<<<dim3((unsigned)M), 256, 0, stream>>>(h, sdP);

    // ---- 10) prediction = succ_sparse . up_W^T (16384 x 512, K=2048) ----
    gemm_bf16_wmma<<<dim3(M / 64, D / 128, 1), 256, 0, stream>>>(
        sdP, uwb, pr, (int)M, D, N, N, N, D, 0, 0, 0);

    // ---- 11) out = x + sigmoid(gl + gate_b) * prediction ----
    {
        const long long total = M * D;
        final_gate_kernel<<<dim3((unsigned)((total + 255) / 256)), 256, 0, stream>>>(
            x, gl, pr, gate_b, out, total, D);
    }
}
